// RotaryCausalAttention_5480378270066
// MI455X (gfx1250) — matrix operand order repaired and checked by reading, not compiled, not run on hardware
//
#include <hip/hip_runtime.h>
#include <hip/hip_bf16.h>

typedef __attribute__((ext_vector_type(16))) _Float16 v16h;
typedef __attribute__((ext_vector_type(8)))  _Float16 v8h;
typedef __attribute__((ext_vector_type(8)))  float    v8f;

#define D_MODEL 1024
#define N_HEADS 16
#define D_HEAD  64
#define SEQ     2048
#define BATCH   2
#define M_TOT   (BATCH * SEQ)   // 4096

__device__ __forceinline__ v16h cat16(v8h lo, v8h hi) {
    return __builtin_shufflevector(lo, hi, 0,1,2,3,4,5,6,7,8,9,10,11,12,13,14,15);
}

__device__ __forceinline__ v8f wmma_f16(v16h a, v16h b, v8f c) {
    return __builtin_amdgcn_wmma_f32_16x16x32_f16(
        /*neg_a=*/false, a, /*neg_b=*/false, b,
        /*c_mod=*/(short)0, c, /*reuse_a=*/false, /*reuse_b=*/false);
}

// ---------------------------------------------------------------------------
// fp32 -> f16 row-major copy (for x staging)
// ---------------------------------------------------------------------------
__global__ void f32_to_f16_kernel(const float* __restrict__ in,
                                  _Float16* __restrict__ out, int n) {
    int i = blockIdx.x * blockDim.x + threadIdx.x;
    if (i < n) out[i] = (_Float16)in[i];
}

// ---------------------------------------------------------------------------
// W[k][n] fp32 -> WT[n][k] f16  (column-major pack so WMMA B-frag loads are
// contiguous 32B per lane)
// ---------------------------------------------------------------------------
__global__ void transpose_w_kernel(const float* __restrict__ W,
                                   _Float16* __restrict__ WT) {
    int t = blockIdx.x * blockDim.x + threadIdx.x;     // 1024*1024 threads
    int n = t & (D_MODEL - 1);
    int k = t >> 10;
    WT[(size_t)n * D_MODEL + k] = (_Float16)W[(size_t)k * D_MODEL + n];
}

// ---------------------------------------------------------------------------
// WMMA GEMM: C[m][n] = sum_k A[m][k] * WT[n][k] + bias[n]
//   mode 0: -> Qh [B,H,S,64] f16, fused partial RoPE
//   mode 1: -> Kh [B,H,S,64] f16, fused partial RoPE
//   mode 2: -> Vt [B,H,64,S] f16 (transposed for PV B-frag loads)
//   mode 3: -> outF [B,S,D_MODEL] fp32 (final projection)
// Block: 256 threads = 8 waves; wave tile 32x64 (2 M-subtiles x 4 N-tiles,
// B-fragments reused across both M-subtiles -> 1.5 loads per WMMA).
// Block tile 256x64.
// ---------------------------------------------------------------------------
__global__ __launch_bounds__(256)
void gemm_kernel(const _Float16* __restrict__ A,
                 const _Float16* __restrict__ BT,
                 const float* __restrict__ bias,
                 _Float16* __restrict__ outH,
                 float* __restrict__ outF,
                 int mode) {
    const int wave = threadIdx.x >> 5;
    const int lane = threadIdx.x & 31;
    const int g    = lane >> 4;          // half-wave group
    const int ln   = lane & 15;
    const int m0   = blockIdx.x * 256 + wave * 32;
    const int n0   = blockIdx.y * 64;

    v8f acc[2][4];
#pragma unroll
    for (int ms = 0; ms < 2; ++ms)
#pragma unroll
        for (int t = 0; t < 4; ++t)
#pragma unroll
            for (int i = 0; i < 8; ++i) acc[ms][t][i] = 0.0f;

    const _Float16* arow0 = A + (size_t)(m0 + ln) * D_MODEL;
    const _Float16* arow1 = A + (size_t)(m0 + 16 + ln) * D_MODEL;

    for (int k0 = 0; k0 < D_MODEL; k0 += 32) {
        if (k0 + 128 < D_MODEL) {
            __builtin_prefetch((const void*)(arow0 + k0 + 128), 0, 1);
            __builtin_prefetch((const void*)(arow1 + k0 + 128), 0, 1);
        }
        // A frags: lane = M; K contiguous [g*16, g*16+16)
        v16h a[2];
        {
            v8h lo0 = *(const v8h*)(arow0 + k0 + g * 16);
            v8h hi0 = *(const v8h*)(arow0 + k0 + 8 + g * 16);
            a[0] = cat16(lo0, hi0);
            v8h lo1 = *(const v8h*)(arow1 + k0 + g * 16);
            v8h hi1 = *(const v8h*)(arow1 + k0 + 8 + g * 16);
            a[1] = cat16(lo1, hi1);
        }
#pragma unroll
        for (int t = 0; t < 4; ++t) {
            // B frag: lane = N col; K contiguous [g*16, g*16+16)
            const _Float16* bp = BT + (size_t)(n0 + t * 16 + ln) * D_MODEL
                                    + k0 + g * 16;
            v16h b = *(const v16h*)bp;
            acc[0][t] = wmma_f16(a[0], b, acc[0][t]);
            acc[1][t] = wmma_f16(a[1], b, acc[1][t]);
        }
    }

    // Epilogue. C layout: VGPR r -> row base + r + g*8; lane%16 -> col.
#pragma unroll
    for (int ms = 0; ms < 2; ++ms) {
#pragma unroll
        for (int t = 0; t < 4; ++t) {
            const int ncol = n0 + t * 16 + ln;
            const float bval = bias[ncol];
            const int h = ncol >> 6;         // head
            const int d = ncol & 63;         // dim within head
#pragma unroll
            for (int r = 0; r < 8; ++r) {
                const int mrow = m0 + ms * 16 + r + g * 8;
                const int b_i  = mrow >> 11;           // batch
                const int s    = mrow & (SEQ - 1);     // token position
                float v = acc[ms][t][r] + bval;
                if (mode <= 1 && t == 0) {
                    // partial RoPE: rotate dims [0,16); pair (d,d+8) via xor-8
                    float pv = __shfl_xor(v, 8);
                    int i = d & 7;
                    float inv = __expf(-9.210340371976184f * 0.125f * (float)i);
                    float ang = (float)s * inv;
                    float sn, cs;
                    __sincosf(ang, &sn, &cs);
                    v = (d < 8) ? (v * cs - pv * sn) : (pv * sn + v * cs);
                }
                if (mode == 0 || mode == 1) {
                    size_t idx = (((size_t)b_i * N_HEADS + h) * SEQ + s) * D_HEAD + d;
                    outH[idx] = (_Float16)v;
                } else if (mode == 2) {
                    size_t idx = (((size_t)b_i * N_HEADS + h) * D_HEAD + d) * SEQ + s;
                    outH[idx] = (_Float16)v;
                } else {
                    outF[(size_t)mrow * D_MODEL + ncol] = v;
                }
            }
        }
    }
}

// ---------------------------------------------------------------------------
// Fused causal flash attention.
// Grid: (S/64 query blocks, B*H). Block: 128 threads = 4 waves.
// Each wave owns 16 query rows; loops over 32-key chunks with online softmax.
// ---------------------------------------------------------------------------
__global__ __launch_bounds__(128)
void attn_kernel(const _Float16* __restrict__ Qh,
                 const _Float16* __restrict__ Kh,
                 const _Float16* __restrict__ Vt,
                 _Float16* __restrict__ attnH) {
    const int w    = threadIdx.x >> 5;
    const int lane = threadIdx.x & 31;
    const int g    = lane >> 4;
    const int ln   = lane & 15;
    const int bh   = blockIdx.y;
    const int q0   = blockIdx.x * 64 + w * 16;

    const _Float16* Qb = Qh + (size_t)bh * SEQ * D_HEAD;
    const _Float16* Kb = Kh + (size_t)bh * SEQ * D_HEAD;
    const _Float16* Vb = Vt + (size_t)bh * D_HEAD * SEQ;

    // Q A-frags (reused for all key chunks): K-dim 64 = two 32-chunks
    const _Float16* qrow = Qb + (size_t)(q0 + ln) * D_HEAD;
    v16h aq[2];
#pragma unroll
    for (int kk = 0; kk < 2; ++kk) {
        v8h lo = *(const v8h*)(qrow + kk * 32 + g * 16);
        v8h hi = *(const v8h*)(qrow + kk * 32 + 8 + g * 16);
        aq[kk] = cat16(lo, hi);
    }

    float mrow[8], lrow[8];
    v8f acc[4];
#pragma unroll
    for (int r = 0; r < 8; ++r) { mrow[r] = -3.0e38f; lrow[r] = 0.0f; }
#pragma unroll
    for (int t = 0; t < 4; ++t)
#pragma unroll
        for (int i = 0; i < 8; ++i) acc[t][i] = 0.0f;

    __shared__ _Float16 plds[4][16 * 40];   // padded P tile per wave
    _Float16* pl = plds[w];

    const int nc = (q0 + 47) >> 5;          // causal: chunks while c*32 <= q0+15
    for (int c = 0; c < nc; ++c) {
        const int kbase = c * 32;

        // --- scores: S = Q @ K^T (two 16-col tiles, K-dim 64) ---
        v8f s[2];
#pragma unroll
        for (int t = 0; t < 2; ++t) {
#pragma unroll
            for (int i = 0; i < 8; ++i) s[t][i] = 0.0f;
            const _Float16* krow = Kb + (size_t)(kbase + t * 16 + ln) * D_HEAD;
#pragma unroll
            for (int kk = 0; kk < 2; ++kk) {
                v16h bk = *(const v16h*)(krow + kk * 32 + g * 16);
                s[t] = wmma_f16(aq[kk], bk, s[t]);
            }
        }

        // --- causal mask + scale + chunk max ---
        float cmax[8];
#pragma unroll
        for (int r = 0; r < 8; ++r) {
            const int row = q0 + r + g * 8;
            float x0 = (kbase + ln      <= row) ? s[0][r] * 0.125f : -3.0e38f;
            float x1 = (kbase + 16 + ln <= row) ? s[1][r] * 0.125f : -3.0e38f;
            s[0][r] = x0; s[1][r] = x1;
            cmax[r] = fmaxf(x0, x1);
        }
        for (int off = 1; off < 16; off <<= 1)
#pragma unroll
            for (int r = 0; r < 8; ++r)
                cmax[r] = fmaxf(cmax[r], __shfl_xor(cmax[r], off));

        // --- online softmax update ---
        float alpha[8], psum[8];
#pragma unroll
        for (int r = 0; r < 8; ++r) {
            float nm = fmaxf(mrow[r], cmax[r]);
            alpha[r] = __expf(mrow[r] - nm);
            mrow[r] = nm;
            float p0 = __expf(s[0][r] - nm);
            float p1 = __expf(s[1][r] - nm);
            psum[r] = p0 + p1;
            pl[(r + g * 8) * 40 + ln]      = (_Float16)p0;
            pl[(r + g * 8) * 40 + 16 + ln] = (_Float16)p1;
        }
        for (int off = 1; off < 16; off <<= 1)
#pragma unroll
            for (int r = 0; r < 8; ++r)
                psum[r] += __shfl_xor(psum[r], off);
#pragma unroll
        for (int r = 0; r < 8; ++r)
            lrow[r] = lrow[r] * alpha[r] + psum[r];
#pragma unroll
        for (int t = 0; t < 4; ++t)
#pragma unroll
            for (int r = 0; r < 8; ++r) acc[t][r] *= alpha[r];

        // wait for P stores before A-frag reload (same-wave LDS RAW)
        asm volatile("s_wait_dscnt 0" ::: "memory");

        // --- P A-frag from LDS (transposed redistribution) ---
        v8h plo = *(const v8h*)(pl + ln * 40 + g * 16);
        v8h phi = *(const v8h*)(pl + ln * 40 + 8 + g * 16);
        v16h ap = cat16(plo, phi);

        // --- acc += P @ V (V stored d-major: contiguous key loads) ---
#pragma unroll
        for (int t = 0; t < 4; ++t) {
            const _Float16* vrow = Vb + (size_t)(t * 16 + ln) * SEQ
                                      + kbase + g * 16;
            v16h bv = *(const v16h*)vrow;
            acc[t] = wmma_f16(ap, bv, acc[t]);
        }
    }

    // --- normalize and store [B,S,D_MODEL] f16 for the O projection ---
    const int b_i = bh >> 4;
    const int h   = bh & 15;
#pragma unroll
    for (int t = 0; t < 4; ++t)
#pragma unroll
        for (int r = 0; r < 8; ++r) {
            const int row = q0 + r + g * 8;
            float o = acc[t][r] / lrow[r];
            size_t idx = ((size_t)b_i * SEQ + row) * D_MODEL
                       + h * D_HEAD + t * 16 + ln;
            attnH[idx] = (_Float16)o;
        }
}

// ---------------------------------------------------------------------------
extern "C" void kernel_launch(void* const* d_in, const int* in_sizes, int n_in,
                              void* d_out, int out_size, void* d_ws, size_t ws_size,
                              hipStream_t stream) {
    const float* x  = (const float*)d_in[0];
    const float* Wq = (const float*)d_in[1];
    const float* bq = (const float*)d_in[2];
    const float* Wk = (const float*)d_in[3];
    const float* bk = (const float*)d_in[4];
    const float* Wv = (const float*)d_in[5];
    const float* bv = (const float*)d_in[6];
    const float* Wo = (const float*)d_in[7];
    const float* bo = (const float*)d_in[8];
    float* out = (float*)d_out;

    char* ws = (char*)d_ws;
    const size_t MB = 1024ull * 1024ull;
    _Float16* xh    = (_Float16*)(ws + 0 * MB);    // 4096x1024   (8 MB)
    _Float16* wtq   = (_Float16*)(ws + 8 * MB);    // 1024x1024 T (2 MB)
    _Float16* wtk   = (_Float16*)(ws + 10 * MB);
    _Float16* wtv   = (_Float16*)(ws + 12 * MB);
    _Float16* wto   = (_Float16*)(ws + 14 * MB);
    _Float16* Qh    = (_Float16*)(ws + 16 * MB);   // [B,H,S,64]  (8 MB)
    _Float16* Kh    = (_Float16*)(ws + 24 * MB);   // [B,H,S,64]  (8 MB)
    _Float16* Vt    = (_Float16*)(ws + 32 * MB);   // [B,H,64,S]  (8 MB)
    _Float16* attnh = (_Float16*)(ws + 40 * MB);   // 4096x1024   (8 MB)

    // 1) pack x to f16
    {
        int n = M_TOT * D_MODEL;
        f32_to_f16_kernel<<<(n + 255) / 256, 256, 0, stream>>>(x, xh, n);
    }
    // 2) transpose-pack all weights to f16 [n][k]
    {
        int blocks = (D_MODEL * D_MODEL) / 256;
        transpose_w_kernel<<<blocks, 256, 0, stream>>>(Wq, wtq);
        transpose_w_kernel<<<blocks, 256, 0, stream>>>(Wk, wtk);
        transpose_w_kernel<<<blocks, 256, 0, stream>>>(Wv, wtv);
        transpose_w_kernel<<<blocks, 256, 0, stream>>>(Wo, wto);
    }
    // 3) QKV projections (RoPE fused into Q/K epilogues)
    dim3 ggrid(M_TOT / 256, D_MODEL / 64);
    gemm_kernel<<<ggrid, 256, 0, stream>>>(xh, wtq, bq, Qh, nullptr, 0);
    gemm_kernel<<<ggrid, 256, 0, stream>>>(xh, wtk, bk, Kh, nullptr, 1);
    gemm_kernel<<<ggrid, 256, 0, stream>>>(xh, wtv, bv, Vt, nullptr, 2);
    // 4) fused causal flash attention
    attn_kernel<<<dim3(SEQ / 64, BATCH * N_HEADS), 128, 0, stream>>>(Qh, Kh, Vt, attnh);
    // 5) output projection -> fp32 d_out
    gemm_kernel<<<ggrid, 256, 0, stream>>>(attnh, wto, bo, nullptr, out, 3);
}
